// Gen3Dmol_Classify_50929722196369
// MI455X (gfx1250) — matrix operand order repaired and checked by reading, not compiled, not run on hardware
//
#include <hip/hip_runtime.h>
#include <hip/hip_bf16.h>
#include <math.h>

// ---------------------------------------------------------------------------
// Gen3Dmol forward on MI455X (gfx1250).
//
// Precision: all GEMMs run bf16 x bf16 -> f32 via v_wmma_f32_16x16x32_bf16
// (wave32). Weights (~190MB fp32) are converted once to bf16 (~95MB), which
// fits in the 192MB L2, so weight streams are L2 hits after first touch.
// The only large HBM stream is the per-layer scores tensor (33.5MB r+w per
// layer -- required: the reference scan feeds layer l's scores in as layer
// l+1's attention bias). Attention is flash-style so probs stay on chip.
//
// All tile-edge handling is branchless (clamped addresses + cndmask zeroing)
// to avoid per-element exec-mask divergence in the staging loops.
// ---------------------------------------------------------------------------

#define BB   4
#define NN   256
#define DD   512
#define HH   32
#define FF   2048
#define LL   15
#define KK   128
#define VV   30
#define HDD  16
#define NEGBIAS (-10000.0f)
#define FSTRIDE 136   // padded row stride for the 129-wide pair features

#define CDIV(a, b) (((a) + (b) - 1) / (b))

typedef __bf16 bf16;
typedef __bf16 bf16x16 __attribute__((ext_vector_type(16)));
typedef float  f32x8   __attribute__((ext_vector_type(8)));

union AFrag { bf16x16 v; uint4 q[2]; bf16 h[16]; };

__device__ __forceinline__ float gelu_f(float x) {
  float x3 = x * x * x;
  return 0.5f * x * (1.0f + tanhf(0.7978845608028654f * (x + 0.044715f * x3)));
}
__device__ __forceinline__ float act_apply(float x, int act) {
  if (act == 1) return gelu_f(x);
  if (act == 2) return tanhf(x);
  return x;
}

// ---------------------------------------------------------------------------
// Generic bf16 WMMA GEMM: out = act(A[MxK] @ W[KxN] + bias) (+ resid)
// Block = 256 threads (8 waves), tile 256x64, K step 32.
// Each wave owns a 32-row strip: 2 A-fragments x 4 B-fragments = 8 WMMAs per
// K-slab. A is staged one full 32-half row per thread (four 16B loads);
// W staged transposed so B-fragments are contiguous 16B ds loads matching
// the CDNA5 B-matrix lane layout. REQUIREMENT: lda % 8 == 0 (all callers
// pad A row strides accordingly) so the vector path is always aligned.
// ---------------------------------------------------------------------------
__global__ __launch_bounds__(256) void gemm_bf16_kernel(
    const bf16* __restrict__ A, int lda,
    const bf16* __restrict__ W, int ldw,
    const float* __restrict__ bias,
    const float* __restrict__ resid,
    float* __restrict__ outF,
    bf16* __restrict__ outB,
    int ldc, int M, int N, int K, int act)
{
  __shared__ __align__(16) bf16 As[256][32];   // 16 KB
  __shared__ __align__(16) bf16 Wt[64][32];    //  4 KB

  const int t    = threadIdx.x;
  const int lane = t & 31;
  const int wave = t >> 5;
  const int m0   = blockIdx.y * 256;
  const int n0   = blockIdx.x * 64;

  f32x8 acc[2][4];
#pragma unroll
  for (int mt = 0; mt < 2; ++mt)
#pragma unroll
    for (int nt = 0; nt < 4; ++nt)
#pragma unroll
      for (int r = 0; r < 8; ++r) acc[mt][nt][r] = 0.0f;

  const int wrow = t >> 3;           // 0..31  (K index within slab)
  const int wcol = (t & 7) * 8;      // 0..56  (N index within tile)

  for (int kt = 0; kt < K; kt += 32) {
    const bool kEdge = (kt + 32 > K);

    // ---- stage A: one 32-half row per thread, branchless edges ----
    {
      int gm  = m0 + t;
      int gmc = (gm < M) ? gm : (M - 1);           // clamp, always valid
      const bf16* src = A + (size_t)gmc * lda + kt;
      if (!kEdge) {
        uint4 u0 = ((const uint4*)src)[0];
        uint4 u1 = ((const uint4*)src)[1];
        uint4 u2 = ((const uint4*)src)[2];
        uint4 u3 = ((const uint4*)src)[3];
        if (gm >= M) {                              // cndmask, no divergence
          u0 = make_uint4(0, 0, 0, 0); u1 = u0; u2 = u0; u3 = u0;
        }
        ((uint4*)As[t])[0] = u0;
        ((uint4*)As[t])[1] = u1;
        ((uint4*)As[t])[2] = u2;
        ((uint4*)As[t])[3] = u3;
        __builtin_prefetch(src + 32, 0, 1);         // next K slab
      } else {
#pragma unroll
        for (int i = 0; i < 32; ++i) {
          int gk  = kt + i;
          int gkc = (gk < K) ? gk : (K - 1);
          bf16 w = src[gkc - kt];                   // unconditional load
          As[t][i] = (gm < M && gk < K) ? w : (bf16)0.0f;
        }
      }
    }

    // ---- stage W transposed (Wt[n][k]), branchless edges ----
    {
      int gk = kt + wrow;
      int gn = n0 + wcol;
      if (((ldw & 7) == 0) && gk < K && (gn + 7) < N) {
        union { uint4 u; bf16 h[8]; } v;
        v.u = *(const uint4*)(W + (size_t)gk * ldw + gn);
#pragma unroll
        for (int i = 0; i < 8; ++i) Wt[wcol + i][wrow] = v.h[i];
      } else {
        int gkc = (gk < K) ? gk : (K - 1);
#pragma unroll
        for (int i = 0; i < 8; ++i) {
          int n  = gn + i;
          int nc = (n < N) ? n : (N - 1);
          bf16 w = W[(size_t)gkc * ldw + nc];       // clamped, unconditional
          Wt[wcol + i][wrow] = (gk < K && n < N) ? w : (bf16)0.0f;
        }
      }
    }
    __syncthreads();

    // ---- 2 A-frags x 4 B-frags -> 8 WMMAs ----
    {
      const int kb  = (lane < 16) ? 0 : 8;     // CDNA5 A layout K base
      const int kb2 = (lane < 16) ? 0 : 8;     // CDNA5 B layout K base
      const int ml  = (wave << 5) + (lane & 15);
      AFrag a0, a1;
      a0.q[0] = *(const uint4*)&As[ml][kb];
      a0.q[1] = *(const uint4*)&As[ml][kb + 16];
      a1.q[0] = *(const uint4*)&As[ml + 16][kb];
      a1.q[1] = *(const uint4*)&As[ml + 16][kb + 16];
#pragma unroll
      for (int nt = 0; nt < 4; ++nt) {
        const int nloc = (nt << 4) + (lane & 15);
        AFrag b;
        b.q[0] = *(const uint4*)&Wt[nloc][kb2];
        b.q[1] = *(const uint4*)&Wt[nloc][kb2 + 16];
        acc[0][nt] = __builtin_amdgcn_wmma_f32_16x16x32_bf16(
            false, a0.v, false, b.v, (short)0, acc[0][nt], false, false);
        acc[1][nt] = __builtin_amdgcn_wmma_f32_16x16x32_bf16(
            false, a1.v, false, b.v, (short)0, acc[1][nt], false, false);
      }
    }
    __syncthreads();
  }

  // ---- epilogue: bias + activation + residual, dual f32/bf16 stores ----
  const int nlane = lane & 15;
#pragma unroll
  for (int mt = 0; mt < 2; ++mt) {
    const int mbase = m0 + (wave << 5) + (mt << 4) + ((lane < 16) ? 0 : 8);
#pragma unroll
    for (int nt = 0; nt < 4; ++nt) {
      int col = n0 + (nt << 4) + nlane;
      if (col >= N) continue;
      float bcol = bias ? bias[col] : 0.0f;
#pragma unroll
      for (int r = 0; r < 8; ++r) {
        int row = mbase + r;
        if (row >= M) continue;
        float v = act_apply(acc[mt][nt][r] + bcol, act);
        if (resid) v += resid[(size_t)row * ldc + col];
        if (outF) outF[(size_t)row * ldc + col] = v;
        if (outB) outB[(size_t)row * ldc + col] = (bf16)v;
      }
    }
  }
}

// ---------------------------------------------------------------------------
// Attention: one block per (b,h), 512 threads = 16 waves, one 16-row q tile
// per wave. HD=16 zero-padded to K=32 for the qk^T WMMA. Flash-style
// streaming softmax over 32-key tiles; probs staged per-wave through LDS to
// convert C-layout -> A-layout for the PV WMMA. scores_out = qk*scale + bias
// is materialized because the reference scan uses it as the next layer bias.
// ---------------------------------------------------------------------------
__global__ __launch_bounds__(512) void attn_kernel(
    const bf16* __restrict__ q, const bf16* __restrict__ k,
    const bf16* __restrict__ v,
    const float* __restrict__ bias_in, float* __restrict__ scores_out,
    bf16* __restrict__ o, float scale)
{
  __shared__ __align__(16) bf16 Ks[32][16];
  __shared__ __align__(16) bf16 Vt[16][32];
  __shared__ __align__(16) bf16 Ps[16][16][40];

  const int bh   = blockIdx.x;
  const int b    = bh / HH;
  const int h    = bh % HH;
  const int t    = threadIdx.x;
  const int lane = t & 31;
  const int wave = t >> 5;          // 0..15
  const int qr0  = wave << 4;

  const int mloc  = lane & 15;
  const int kb    = (lane < 16) ? 0 : 8;
  const int nlane = lane & 15;
  const int moff  = (lane < 16) ? 0 : 8;

  AFrag qa;
  {
    const bf16* qp =
        q + (((size_t)(b * NN + qr0 + mloc) * HH + h) * HDD) + kb;
    qa.q[0] = *(const uint4*)qp;       // d = kb..kb+7 (real)
    qa.q[1] = make_uint4(0, 0, 0, 0);  // d >= 16 padding
  }

  float mrow[8], lrow[8];
  f32x8 oacc;
#pragma unroll
  for (int r = 0; r < 8; ++r) { mrow[r] = -1e30f; lrow[r] = 0.0f; oacc[r] = 0.0f; }

  for (int kt = 0; kt < NN; kt += 32) {
    {   // cooperative K/V tile staging (coalesced global, 1 elem/thread)
      int key = t >> 4;
      int d   = t & 15;
      size_t base = ((size_t)(b * NN + kt + key) * HH + h) * HDD + d;
      Ks[key][d] = k[base];
      Vt[d][key] = v[base];
    }
    __syncthreads();

    f32x8 s0, s1;
    {
      AFrag b0, b1;
      b0.q[0] = *(const uint4*)&Ks[nlane][kb];
      b0.q[1] = make_uint4(0, 0, 0, 0);
      b1.q[0] = *(const uint4*)&Ks[16 + nlane][kb];
      b1.q[1] = make_uint4(0, 0, 0, 0);
      f32x8 z;
#pragma unroll
      for (int r = 0; r < 8; ++r) z[r] = 0.0f;
      s0 = __builtin_amdgcn_wmma_f32_16x16x32_bf16(false, qa.v, false, b0.v,
                                                   (short)0, z, false, false);
      s1 = __builtin_amdgcn_wmma_f32_16x16x32_bf16(false, qa.v, false, b1.v,
                                                   (short)0, z, false, false);
    }

    // bias + scale, score materialization, streaming softmax update
#pragma unroll
    for (int r = 0; r < 8; ++r) {
      int row = qr0 + moff + r;
      size_t base = (((size_t)(b * HH + h) * NN) + row) * NN + kt;
      float v0 = s0[r] * scale + bias_in[base + nlane];
      float v1 = s1[r] * scale + bias_in[base + 16 + nlane];
      scores_out[base + nlane]      = v0;
      scores_out[base + 16 + nlane] = v1;
      float mx = fmaxf(v0, v1);
#pragma unroll
      for (int off = 8; off; off >>= 1) mx = fmaxf(mx, __shfl_xor(mx, off, 32));
      float mnew  = fmaxf(mrow[r], mx);
      float alpha = __expf(mrow[r] - mnew);
      float p0 = __expf(v0 - mnew);
      float p1 = __expf(v1 - mnew);
      float rs = p0 + p1;
#pragma unroll
      for (int off = 8; off; off >>= 1) rs += __shfl_xor(rs, off, 32);
      lrow[r] = lrow[r] * alpha + rs;
      mrow[r] = mnew;
      oacc[r] *= alpha;
      Ps[wave][moff + r][nlane]      = (bf16)p0;
      Ps[wave][moff + r][16 + nlane] = (bf16)p1;
    }

    {   // oacc += P(16x32) @ V(32x16)
      AFrag pf, vf;
      pf.q[0] = *(const uint4*)&Ps[wave][mloc][kb];
      pf.q[1] = *(const uint4*)&Ps[wave][mloc][kb + 16];
      const int kb2 = (lane < 16) ? 0 : 8;
      vf.q[0] = *(const uint4*)&Vt[nlane][kb2];
      vf.q[1] = *(const uint4*)&Vt[nlane][kb2 + 16];
      oacc = __builtin_amdgcn_wmma_f32_16x16x32_bf16(
          false, pf.v, false, vf.v, (short)0, oacc, false, false);
    }
    __syncthreads();
  }

#pragma unroll
  for (int r = 0; r < 8; ++r) {
    int row = qr0 + moff + r;
    float inv = 1.0f / lrow[r];
    o[(((size_t)(b * NN + row) * HH) + h) * HDD + nlane] = (bf16)(oacc[r] * inv);
  }
}

// ---------------------------------------------------------------------------
// LayerNorm over rows of length D (block per row, 256 threads).
// ---------------------------------------------------------------------------
__global__ __launch_bounds__(256) void ln_kernel(
    const float* __restrict__ x, const float* __restrict__ g,
    const float* __restrict__ bvec, float* __restrict__ outF,
    bf16* __restrict__ outB, int D)
{
  __shared__ float sm[8];
  const size_t row = blockIdx.x;
  const float* xr = x + row * D;

  float s = 0.0f;
  for (int i = threadIdx.x; i < D; i += 256) s += xr[i];
#pragma unroll
  for (int off = 16; off; off >>= 1) s += __shfl_xor(s, off, 32);
  if ((threadIdx.x & 31) == 0) sm[threadIdx.x >> 5] = s;
  __syncthreads();
  float mean = 0.0f;
#pragma unroll
  for (int i = 0; i < 8; ++i) mean += sm[i];
  mean /= (float)D;
  __syncthreads();

  float vs = 0.0f;
  for (int i = threadIdx.x; i < D; i += 256) {
    float d = xr[i] - mean;
    vs += d * d;
  }
#pragma unroll
  for (int off = 16; off; off >>= 1) vs += __shfl_xor(vs, off, 32);
  if ((threadIdx.x & 31) == 0) sm[threadIdx.x >> 5] = vs;
  __syncthreads();
  float var = 0.0f;
#pragma unroll
  for (int i = 0; i < 8; ++i) var += sm[i];
  var /= (float)D;
  float rstd = rsqrtf(var + 1e-5f);

  for (int i = threadIdx.x; i < D; i += 256) {
    float y = (xr[i] - mean) * rstd * g[i] + bvec[i];
    if (outF) outF[row * D + i] = y;
    if (outB) outB[row * D + i] = (bf16)y;
  }
}

// ---------------------------------------------------------------------------
// Small elementwise / feature kernels.
// ---------------------------------------------------------------------------
__global__ void cvt_bf16_kernel(const float* __restrict__ s,
                                bf16* __restrict__ d, int n) {
  int i = blockIdx.x * blockDim.x + threadIdx.x;
  if (i < n) d[i] = (bf16)s[i];
}

__global__ void embed_kernel(const int* __restrict__ tok,
                             const int* __restrict__ iT,
                             const float* __restrict__ press,
                             const float* __restrict__ temp,
                             const float* __restrict__ embed_tokens,
                             const float* __restrict__ tok_time,
                             const float* __restrict__ tok_pw,
                             const float* __restrict__ tok_pb,
                             const float* __restrict__ tok_tw,
                             const float* __restrict__ tok_tb,
                             float* __restrict__ x) {
  int idx = blockIdx.x * blockDim.x + threadIdx.x;
  if (idx >= BB * NN * DD) return;
  int d  = idx % DD;
  int bn = idx / DD;
  int b  = bn / NN;
  int token = tok[bn];
  x[idx] = embed_tokens[token * DD + d] + tok_time[iT[b] * DD + d] +
           press[b] * tok_pw[d] + tok_pb[d] + temp[b] * tok_tw[d] + tok_tb[d];
}

__global__ void emb_bias_kernel(const int* __restrict__ iT,
                                const float* __restrict__ press,
                                const float* __restrict__ temp,
                                const float* __restrict__ ebb_time,
                                const float* __restrict__ pw,
                                const float* __restrict__ pb,
                                const float* __restrict__ tw,
                                const float* __restrict__ tb,
                                float* __restrict__ out) {
  int idx = threadIdx.x;
  if (idx >= BB * HH) return;
  int b = idx / HH, h = idx % HH;
  out[idx] = ebb_time[iT[b] * HH + h] + press[b] * pw[h] + pb[h] +
             temp[b] * tw[h] + tb[h];
}

__global__ __launch_bounds__(160) void gbf_feat_kernel(
    const int* __restrict__ tok, const float* __restrict__ coord,
    const float* __restrict__ means, const float* __restrict__ stds,
    const float* __restrict__ mul, const float* __restrict__ bia,
    bf16* __restrict__ feat) {
  int pair = blockIdx.x;
  int c = threadIdx.x;
  if (c >= KK + 1) return;
  int b  = pair >> 16;           // N*N = 65536
  int ij = pair & 65535;
  int i = ij >> 8, j = ij & 255;
  const float* ci = coord + (size_t)(b * NN + i) * 3;
  const float* cj = coord + (size_t)(b * NN + j) * 3;
  float dx = ci[0] - cj[0], dy = ci[1] - cj[1], dz = ci[2] - cj[2];
  float d2 = dx * dx + dy * dy + dz * dz;
  float dist = (d2 > 0.0f) ? sqrtf(d2) : 0.0f;
  float val;
  if (c < KK) {
    int et = tok[b * NN + i] * VV + tok[b * NN + j];
    float xg = mul[et] * dist + bia[et];
    float sd = fabsf(stds[c]) + 1e-5f;
    float a = (xg - means[c]) / sd;
    val = __expf(-0.5f * a * a) / (2.5066282746310002f * sd);
  } else {
    val = dist;
  }
  feat[(size_t)pair * FSTRIDE + c] = (bf16)val;
}

__global__ void bias_finalize_kernel(const float* __restrict__ g2,
                                     const float* __restrict__ embb,
                                     const int* __restrict__ tok,
                                     float* __restrict__ bias0) {
  size_t idx = (size_t)blockIdx.x * blockDim.x + threadIdx.x;
  size_t total = (size_t)BB * HH * NN * NN;
  if (idx >= total) return;
  int j = idx % NN;
  size_t r = idx / NN;
  int i = r % NN; r /= NN;
  int h = r % HH;
  int b = (int)(r / HH);
  float val = (tok[b * NN + j] == 0)
                  ? NEGBIAS
                  : g2[(((size_t)(b * NN + i) * NN) + j) * HH + h] +
                        embb[b * HH + h];
  bias0[idx] = val;
}

__global__ void delta_pair_kernel(const float* __restrict__ scores,
                                  const float* __restrict__ bias0,
                                  const int* __restrict__ tok,
                                  bf16* __restrict__ dp) {
  size_t idx = (size_t)blockIdx.x * blockDim.x + threadIdx.x;
  size_t total = (size_t)BB * NN * NN * HH;
  if (idx >= total) return;
  int h = idx % HH;
  size_t r = idx / HH;
  int j = r % NN; r /= NN;
  int i = r % NN;
  int b = (int)(r / NN);
  float val = 0.0f;
  if (tok[b * NN + j] != 0) {
    size_t s = (((size_t)(b * HH + h) * NN) + i) * NN + j;
    val = scores[s] - bias0[s];
  }
  dp[idx] = (bf16)val;
}

__global__ __launch_bounds__(256) void atomnum_kernel(
    const int* __restrict__ tok, float* __restrict__ an) {
  __shared__ float sm[8];
  int b = blockIdx.x;
  float c = (tok[b * NN + threadIdx.x] != 0) ? 1.0f : 0.0f;
#pragma unroll
  for (int off = 16; off; off >>= 1) c += __shfl_xor(c, off, 32);
  if ((threadIdx.x & 31) == 0) sm[threadIdx.x >> 5] = c;
  __syncthreads();
  if (threadIdx.x == 0) {
    float tot = 0.0f;
    for (int w = 0; w < 8; ++w) tot += sm[w];
    an[b] = tot - 3.0f;
  }
}

__global__ __launch_bounds__(256) void coord_update_kernel(
    const float* __restrict__ coord, const float* __restrict__ ap,
    const float* __restrict__ an, float* __restrict__ cu,
    bf16* __restrict__ cub) {
  __shared__ float sm[3][8];
  int bi = blockIdx.x;
  int b = bi / NN, i = bi % NN;
  int j = threadIdx.x;
  float a = ap[(size_t)bi * NN + j];
  const float* ci = coord + (size_t)(b * NN + i) * 3;
  const float* cj = coord + (size_t)(b * NN + j) * 3;
  float s[3];
  s[0] = (cj[0] - ci[0]) * a;   // delta_pos = -diff
  s[1] = (cj[1] - ci[1]) * a;
  s[2] = (cj[2] - ci[2]) * a;
#pragma unroll
  for (int d = 0; d < 3; ++d) {
    float v = s[d];
#pragma unroll
    for (int off = 16; off; off >>= 1) v += __shfl_xor(v, off, 32);
    if ((threadIdx.x & 31) == 0) sm[d][threadIdx.x >> 5] = v;
  }
  __syncthreads();
  if (threadIdx.x == 0) {
    float inv = 1.0f / an[b];
#pragma unroll
    for (int d = 0; d < 3; ++d) {
      float tot = 0.0f;
      for (int w = 0; w < 8; ++w) tot += sm[d][w];
      float r = tot * inv;
      cu[(size_t)bi * 3 + d] = r;
      cub[(size_t)bi * 8 + d] = (bf16)r;
    }
#pragma unroll
    for (int d = 3; d < 8; ++d) cub[(size_t)bi * 8 + d] = (bf16)0.0f;
  }
}

// ---------------------------------------------------------------------------
// Host side
// ---------------------------------------------------------------------------
static void gemm(hipStream_t s, const bf16* A, int lda, const bf16* W, int ldw,
                 const float* bias, const float* resid, float* outF,
                 bf16* outB, int ldc, int M, int N, int K, int act) {
  dim3 g(CDIV(N, 64), CDIV(M, 256));
  gemm_bf16_kernel<<<g, 256, 0, s>>>(A, lda, W, ldw, bias, resid, outF, outB,
                                     ldc, M, N, K, act);
}
static void cvt(hipStream_t s, const float* src, bf16* dst, int n) {
  cvt_bf16_kernel<<<CDIV(n, 256), 256, 0, s>>>(src, dst, n);
}

extern "C" void kernel_launch(void* const* d_in, const int* in_sizes, int n_in,
                              void* d_out, int out_size, void* d_ws,
                              size_t ws_size, hipStream_t stream) {
  (void)in_sizes; (void)n_in; (void)out_size; (void)ws_size;
  // Input order: depth-first insertion order of the setup_inputs() dict.
  auto fin = [&](int i) { return (const float*)d_in[i]; };
  auto iin = [&](int i) { return (const int*)d_in[i]; };

  const int64_t BN = (int64_t)BB * NN;            // 1024
  const int64_t MP = (int64_t)BB * NN * NN;       // 262144

  char* wsb = (char*)d_ws;
  size_t off = 0;
  auto alloc = [&](size_t bytes) -> void* {
    void* p = wsb + off;
    off += (bytes + 255) & ~(size_t)255;
    return p;
  };

  // bf16 weight copies (fit in L2 collectively)
  bf16* wq   = (bf16*)alloc((size_t)LL * DD * DD * 2);
  bf16* wk   = (bf16*)alloc((size_t)LL * DD * DD * 2);
  bf16* wv   = (bf16*)alloc((size_t)LL * DD * DD * 2);
  bf16* wo   = (bf16*)alloc((size_t)LL * DD * DD * 2);
  bf16* w1m  = (bf16*)alloc((size_t)LL * DD * FF * 2);
  bf16* w2m  = (bf16*)alloc((size_t)LL * FF * DD * 2);
  bf16* gp1w1 = (bf16*)alloc(129 * 129 * 2);
  bf16* gp1w2 = (bf16*)alloc(129 * HH * 2);
  bf16* gp2w1 = (bf16*)alloc(HH * HH * 2);
  bf16* gp2w2 = (bf16*)alloc(HH * HH * 2);
  bf16* lmw    = (bf16*)alloc(DD * DD * 2);
  bf16* lmoutw = (bf16*)alloc(DD * 10 * 2);
  bf16* p2cw1  = (bf16*)alloc(HH * HH * 2);
  bf16* p2cw2  = (bf16*)alloc(HH * 2);
  bf16* clsw    = (bf16*)alloc(DD * DD * 2);
  bf16* clsoutw = (bf16*)alloc(DD * 3 * 2);
  bf16* tmpw    = (bf16*)alloc(DD * DD * 2);
  bf16* tmpoutw = (bf16*)alloc(DD * 2);
  bf16* prsw    = (bf16*)alloc(DD * DD * 2);
  bf16* prsoutw = (bf16*)alloc(DD * 2);
  bf16* unw1 = (bf16*)alloc(3 * FF * 2);
  bf16* unw2 = (bf16*)alloc(FF * 3 * 2);

  // activations / intermediates
  bf16*  feat  = (bf16*)alloc((size_t)MP * FSTRIDE * 2);
  bf16*  g1    = (bf16*)alloc((size_t)MP * FSTRIDE * 2);
  bf16*  ga    = (bf16*)alloc((size_t)MP * HH * 2);
  bf16*  gbuf  = (bf16*)alloc((size_t)MP * HH * 2);
  float* g2f   = (float*)alloc((size_t)MP * HH * 4);
  float* bias0 = (float*)alloc((size_t)BB * HH * NN * NN * 4);
  float* scoresb = (float*)alloc((size_t)BB * HH * NN * NN * 4);
  float* embb  = (float*)alloc(BB * HH * 4);
  float* x     = (float*)alloc((size_t)BN * DD * 4);
  bf16*  hbf   = (bf16*)alloc((size_t)BN * DD * 2);
  bf16*  qbuf  = (bf16*)alloc((size_t)BN * DD * 2);
  bf16*  kbuf  = (bf16*)alloc((size_t)BN * DD * 2);
  bf16*  vbuf  = (bf16*)alloc((size_t)BN * DD * 2);
  bf16*  obuf  = (bf16*)alloc((size_t)BN * DD * 2);
  bf16*  f1bf  = (bf16*)alloc((size_t)BN * FF * 2);
  float* xlnf  = (float*)alloc((size_t)BN * DD * 4);
  bf16*  xlnbf = (bf16*)alloc((size_t)BN * DD * 2);
  float* lm1f  = (float*)alloc((size_t)BN * DD * 4);
  bf16*  hhbf  = (bf16*)alloc((size_t)BN * DD * 2);
  bf16*  dp    = (bf16*)alloc((size_t)MP * HH * 2);
  bf16*  pc1   = (bf16*)alloc((size_t)MP * HH * 2);
  float* apf   = (float*)alloc((size_t)MP * 4);
  float* anum  = (float*)alloc(BB * 4);
  float* cuf   = (float*)alloc((size_t)BN * 3 * 4);
  bf16*  cub   = (bf16*)alloc((size_t)BN * 8 * 2);
  bf16*  un1   = (bf16*)alloc((size_t)BN * FF * 2);
  bf16*  cls1  = (bf16*)alloc((size_t)BB * DD * 2);
  bf16*  tmp1  = (bf16*)alloc((size_t)BB * DD * 2);
  bf16*  prs1  = (bf16*)alloc((size_t)BB * DD * 2);

  const int* tok   = iin(69);
  const float* coord = fin(70);
  const int* iT    = iin(71);
  const float* press = fin(72);
  const float* temp  = fin(73);
  float* out = (float*)d_out;

  // ---- weight conversion fp32 -> bf16 ----
  cvt(stream, fin(5),  wq,  LL * DD * DD);
  cvt(stream, fin(7),  wk,  LL * DD * DD);
  cvt(stream, fin(9),  wv,  LL * DD * DD);
  cvt(stream, fin(11), wo,  LL * DD * DD);
  cvt(stream, fin(17), w1m, LL * DD * FF);
  cvt(stream, fin(19), w2m, LL * FF * DD);
  cvt(stream, fin(25), gp1w1, 129 * 129);
  cvt(stream, fin(27), gp1w2, 129 * HH);
  cvt(stream, fin(29), gp2w1, HH * HH);
  cvt(stream, fin(31), gp2w2, HH * HH);
  cvt(stream, fin(43), lmw, DD * DD);
  cvt(stream, fin(47), lmoutw, DD * 10);
  cvt(stream, fin(49), p2cw1, HH * HH);
  cvt(stream, fin(51), p2cw2, HH);
  cvt(stream, fin(53), clsw, DD * DD);
  cvt(stream, fin(55), clsoutw, DD * 3);
  cvt(stream, fin(57), tmpw, DD * DD);
  cvt(stream, fin(59), tmpoutw, DD);
  cvt(stream, fin(61), prsw, DD * DD);
  cvt(stream, fin(63), prsoutw, DD);
  cvt(stream, fin(65), unw1, 3 * FF);
  cvt(stream, fin(67), unw2, FF * 3);

  // ---- pairwise GBF bias network ----
  gbf_feat_kernel<<<(unsigned)MP, 160, 0, stream>>>(
      tok, coord, fin(21), fin(22), fin(23), fin(24), feat);
  gemm(stream, feat, FSTRIDE, gp1w1, 129, fin(26), nullptr, nullptr, g1,
       FSTRIDE, (int)MP, 129, 129, 1);
  gemm(stream, g1, FSTRIDE, gp1w2, HH, fin(28), nullptr, nullptr, ga, HH,
       (int)MP, HH, 129, 0);
  gemm(stream, ga, HH, gp2w1, HH, fin(30), nullptr, nullptr, gbuf, HH,
       (int)MP, HH, HH, 1);
  gemm(stream, gbuf, HH, gp2w2, HH, fin(32), nullptr, g2f, nullptr, HH,
       (int)MP, HH, HH, 0);
  emb_bias_kernel<<<1, 128, 0, stream>>>(iT, press, temp, fin(33), fin(34),
                                         fin(35), fin(36), fin(37), embb);
  bias_finalize_kernel<<<CDIV((unsigned)(BB * HH * NN * NN), 256), 256, 0,
                         stream>>>(g2f, embb, tok, bias0);

  // ---- embedding + emb LN ----
  embed_kernel<<<CDIV((unsigned)(BN * DD), 256), 256, 0, stream>>>(
      tok, iT, press, temp, fin(0), fin(38), fin(39), fin(40), fin(41),
      fin(42), x);
  ln_kernel<<<(unsigned)BN, 256, 0, stream>>>(x, fin(1), fin(2), x, nullptr,
                                              DD);

  // ---- transformer layers (bias for layer l = scores of layer l-1) ----
  const float scale = 0.25f;   // HD^-0.5, HD=16
  for (int l = 0; l < LL; ++l) {
    const bf16* wql = wq  + (size_t)l * DD * DD;
    const bf16* wkl = wk  + (size_t)l * DD * DD;
    const bf16* wvl = wv  + (size_t)l * DD * DD;
    const bf16* wol = wo  + (size_t)l * DD * DD;
    const bf16* w1l = w1m + (size_t)l * DD * FF;
    const bf16* w2l = w2m + (size_t)l * FF * DD;
    const float* bq = fin(6)  + (size_t)l * DD;
    const float* bk = fin(8)  + (size_t)l * DD;
    const float* bv = fin(10) + (size_t)l * DD;
    const float* bo = fin(12) + (size_t)l * DD;
    const float* b1 = fin(18) + (size_t)l * FF;
    const float* b2 = fin(20) + (size_t)l * DD;

    ln_kernel<<<(unsigned)BN, 256, 0, stream>>>(
        x, fin(13) + (size_t)l * DD, fin(14) + (size_t)l * DD, nullptr, hbf,
        DD);
    gemm(stream, hbf, DD, wql, DD, bq, nullptr, nullptr, qbuf, DD, (int)BN,
         DD, DD, 0);
    gemm(stream, hbf, DD, wkl, DD, bk, nullptr, nullptr, kbuf, DD, (int)BN,
         DD, DD, 0);
    gemm(stream, hbf, DD, wvl, DD, bv, nullptr, nullptr, vbuf, DD, (int)BN,
         DD, DD, 0);
    attn_kernel<<<BB * HH, 512, 0, stream>>>(
        qbuf, kbuf, vbuf, (l == 0) ? bias0 : scoresb, scoresb, obuf, scale);
    gemm(stream, obuf, DD, wol, DD, bo, x, x, nullptr, DD, (int)BN, DD, DD,
         0);
    ln_kernel<<<(unsigned)BN, 256, 0, stream>>>(
        x, fin(15) + (size_t)l * DD, fin(16) + (size_t)l * DD, nullptr, hbf,
        DD);
    gemm(stream, hbf, DD, w1l, FF, b1, nullptr, nullptr, f1bf, FF, (int)BN,
         FF, DD, 1);
    gemm(stream, f1bf, FF, w2l, DD, b2, x, x, nullptr, DD, (int)BN, DD, FF,
         0);
  }

  // ---- final LN ----
  ln_kernel<<<(unsigned)BN, 256, 0, stream>>>(x, fin(3), fin(4), xlnf, xlnbf,
                                              DD);

  // ---- LM head: logits -> out[0 .. 10240) ----
  gemm(stream, xlnbf, DD, lmw, DD, fin(44), nullptr, lm1f, nullptr, DD,
       (int)BN, DD, DD, 1);
  ln_kernel<<<(unsigned)BN, 256, 0, stream>>>(lm1f, fin(45), fin(46), nullptr,
                                              hhbf, DD);
  gemm(stream, hhbf, DD, lmoutw, 10, fin(48), nullptr, out, nullptr, 10,
       (int)BN, 10, DD, 0);

  // ---- delta-pair / p2c / coord update / up_noise -> out[10260..13332) ----
  delta_pair_kernel<<<CDIV((unsigned)(MP * HH), 256), 256, 0, stream>>>(
      scoresb, bias0, tok, dp);
  gemm(stream, dp, HH, p2cw1, HH, fin(50), nullptr, nullptr, pc1, HH,
       (int)MP, HH, HH, 1);
  gemm(stream, pc1, HH, p2cw2, 1, fin(52), nullptr, apf, nullptr, 1, (int)MP,
       1, HH, 0);
  atomnum_kernel<<<BB, 256, 0, stream>>>(tok, anum);
  coord_update_kernel<<<(unsigned)BN, 256, 0, stream>>>(coord, apf, anum, cuf,
                                                        cub);
  gemm(stream, cub, 8, unw1, FF, fin(66), nullptr, nullptr, un1, FF, (int)BN,
       FF, 3, 1);
  gemm(stream, un1, FF, unw2, 3, fin(68), nullptr, out + 10260, nullptr, 3,
       (int)BN, 3, FF, 0);

  // ---- cls heads (rows b*N of xlnbf via lda = N*D) ----
  gemm(stream, xlnbf, NN * DD, clsw, DD, fin(54), nullptr, nullptr, cls1, DD,
       BB, DD, DD, 2);
  gemm(stream, cls1, DD, clsoutw, 3, fin(56), nullptr, out + 10240, nullptr,
       3, BB, 3, DD, 0);
  gemm(stream, xlnbf, NN * DD, tmpw, DD, fin(58), nullptr, nullptr, tmp1, DD,
       BB, DD, DD, 2);
  gemm(stream, tmp1, DD, tmpoutw, 1, fin(60), nullptr, out + 10252, nullptr,
       1, BB, 1, DD, 0);
  gemm(stream, xlnbf, NN * DD, prsw, DD, fin(62), nullptr, nullptr, prs1, DD,
       BB, DD, DD, 2);
  gemm(stream, prs1, DD, prsoutw, 1, fin(64), nullptr, out + 10256, nullptr,
       1, BB, 1, DD, 0);
}
